// GAT_3633542332518
// MI455X (gfx1250) — matrix operand order repaired and checked by reading, not compiled, not run on hardware
//
#include <hip/hip_runtime.h>
#include <hip/hip_bf16.h>

#define N_NODES 50000
#define E_EDGES 800000
#define IN_F    512
#define HD      256          // H*D
#define NEG_SLOPE 0.2f

typedef __attribute__((ext_vector_type(16))) __bf16 v16bf;
typedef __attribute__((ext_vector_type(8)))  float  v8f;

union ABFrag { v16bf v; unsigned int u[8]; };

// monotone float->uint key for atomicMax-based segment max
__device__ __forceinline__ unsigned int fkey(float f) {
  unsigned int b = __float_as_uint(f);
  return (b & 0x80000000u) ? ~b : (b | 0x80000000u);
}
__device__ __forceinline__ float fkey_inv(unsigned int u) {
  unsigned int b = (u & 0x80000000u) ? (u ^ 0x80000000u) : ~u;
  return __uint_as_float(b);
}

// ---------------- conversion / prep kernels ----------------

__global__ void k_cvt_bf16(const float* __restrict__ in, __bf16* __restrict__ out, long n) {
  long i = (long)blockIdx.x * blockDim.x + threadIdx.x;
  long stride = (long)gridDim.x * blockDim.x;
  for (; i < n; i += stride) out[i] = (__bf16)in[i];
}

// W [K, Ncols] fp32 row-major -> Wt rows [rowOff .. rowOff+Ncols) of a [*, K] bf16 buffer
__global__ void k_cvt_w_t(const float* __restrict__ W, __bf16* __restrict__ Wt,
                          int K, int Ncols, int rowOff) {
  int t = blockIdx.x * blockDim.x + threadIdx.x;
  int total = K * Ncols;
  if (t >= total) return;
  int k = t / Ncols, c = t % Ncols;
  Wt[(size_t)(rowOff + c) * K + k] = (__bf16)W[t];
}

__global__ void k_bias_combine(const float* __restrict__ bs, const float* __restrict__ bd,
                               float* __restrict__ bcomb) {
  int t = blockIdx.x * blockDim.x + threadIdx.x;
  if (t >= 768) return;
  float v = 0.0f;
  if (t < 256) v = bs[t];
  else if (t < 512) v = bd[t - 256];
  bcomb[t] = v;
}

// ---------------- WMMA bf16 GEMM:  C[M,Ncols] = A[M,K] * Bt[Ncols,K]^T + bias ----------------
// block = 128 threads (4 waves), block tile = 64(M) x 64(N), wave tile = 16(M) x 64(N)
__global__ void k_gemm_bf16(const __bf16* __restrict__ A, const __bf16* __restrict__ Bt,
                            const float* __restrict__ bias, float* __restrict__ C,
                            int M, int Ncols, int K, int relu) {
  const int lane = threadIdx.x & 31;
  const int wave = threadIdx.x >> 5;
  const int mBase = blockIdx.y * 64 + wave * 16;
  const int nBase = blockIdx.x * 64;
  if (mBase >= M) return;

  const int ml   = lane & 15;
  const int half = lane >> 4;
  const int aRow = mBase + ml;          // A row this lane feeds (M = lane%16)

  v8f acc[4];
  #pragma unroll
  for (int i = 0; i < 4; ++i) acc[i] = (v8f){};

  for (int k0 = 0; k0 < K; k0 += 32) {
    // A fragment: 16x32 bf16. VGPR v: K = half*16 + 2v, 2v+1
    ABFrag a;
    const __bf16* ap = A + (size_t)aRow * K + k0;
    #pragma unroll
    for (int v = 0; v < 8; ++v) {
      int kb = half * 16 + 2 * v;
      a.u[v] = *(const unsigned int*)(ap + kb);
    }
    #pragma unroll
    for (int nb = 0; nb < 4; ++nb) {
      // B fragment: 32x16 bf16, lane = N, VGPR j: K = half*16 + 2j, 2j+1 (contiguous in Bt)
      ABFrag b;
      const __bf16* bp = Bt + (size_t)(nBase + nb * 16 + ml) * K + k0 + half * 16;
      #pragma unroll
      for (int j = 0; j < 8; ++j) b.u[j] = *(const unsigned int*)(bp + 2 * j);
      acc[nb] = __builtin_amdgcn_wmma_f32_16x16x32_bf16(
          false, a.v, false, b.v, (short)0, acc[nb], false, false);
    }
  }

  // C/D layout: lane%16 = N, VGPR r -> M = half*8 + r
  #pragma unroll
  for (int nb = 0; nb < 4; ++nb) {
    int ncol = nBase + nb * 16 + ml;
    float bv = bias ? bias[ncol] : 0.0f;
    #pragma unroll
    for (int r = 0; r < 8; ++r) {
      int mrow = mBase + half * 8 + r;
      float val = acc[nb][r] + bv;
      if (relu) val = fmaxf(val, 0.0f);
      C[(size_t)mrow * Ncols + ncol] = val;
    }
  }
}

// ---------------- edge phase ----------------
// F layout: [N, 768] = [fs(0:256) | fd(256:512) | res(512:768)]

// one wave per edge: logits[e,h] = sum_d attn[h,d]*leakyrelu(fs[src]+fd[dst]); atomicMax seg-max
__global__ void k_edge_logits(const float* __restrict__ F, const int* __restrict__ src,
                              const int* __restrict__ dst, const float* __restrict__ attn,
                              float* __restrict__ logits, unsigned int* __restrict__ mkey) {
  int wid  = (int)(((long)blockIdx.x * blockDim.x + threadIdx.x) >> 5);
  int lane = threadIdx.x & 31;
  if (wid >= E_EDGES) return;
  int s = src[wid], d = dst[wid];

  const float4* fs = (const float4*)(F + (size_t)s * 768 + lane * 8);
  const float4* fd = (const float4*)(F + (size_t)d * 768 + 256 + lane * 8);
  const float4* at = (const float4*)(attn + lane * 8);
  float u[8], v[8], aw[8];
  *(float4*)&u[0]  = fs[0]; *(float4*)&u[4]  = fs[1];
  *(float4*)&v[0]  = fd[0]; *(float4*)&v[4]  = fd[1];
  *(float4*)&aw[0] = at[0]; *(float4*)&aw[4] = at[1];

  float sum = 0.0f;
  #pragma unroll
  for (int j = 0; j < 8; ++j) {
    float t = u[j] + v[j];
    sum += aw[j] * (t > 0.0f ? t : NEG_SLOPE * t);
  }
  // reduce within each 8-lane group (one head per group)
  sum += __shfl_xor(sum, 1, 32);
  sum += __shfl_xor(sum, 2, 32);
  sum += __shfl_xor(sum, 4, 32);

  if ((lane & 7) == 0) {
    int h = lane >> 3;
    logits[(size_t)wid * 4 + h] = sum;
    atomicMax(&mkey[(size_t)d * 4 + h], fkey(sum));
  }
}

// a = exp(logit - m[dst]); denom[dst] += a; aexp stored in-place over logits
__global__ void k_exp_denom(const int* __restrict__ dst, float* __restrict__ logits,
                            const unsigned int* __restrict__ mkey, float* __restrict__ denom) {
  long t = (long)blockIdx.x * blockDim.x + threadIdx.x;
  if (t >= (long)E_EDGES * 4) return;
  int e = (int)(t >> 2), h = (int)(t & 3);
  int d = dst[e];
  float m = fkey_inv(mkey[(size_t)d * 4 + h]);
  // guard empty/non-finite segment max (matches reference isfinite guard)
  if ((__float_as_uint(m) & 0x7F800000u) == 0x7F800000u) m = 0.0f;
  float a = __expf(logits[t] - m);
  logits[t] = a;
  atomicAdd(&denom[(size_t)d * 4 + h], a);
}

// one wave per edge: rst[dst] += (a/denom) * fs[src]
__global__ void k_aggregate(const float* __restrict__ F, const int* __restrict__ src,
                            const int* __restrict__ dst, const float* __restrict__ aexp,
                            const float* __restrict__ denom, float* __restrict__ rst) {
  int wid  = (int)(((long)blockIdx.x * blockDim.x + threadIdx.x) >> 5);
  int lane = threadIdx.x & 31;
  if (wid >= E_EDGES) return;
  int s = src[wid], d = dst[wid];
  int h = lane >> 3;
  float alpha = aexp[(size_t)wid * 4 + h] / denom[(size_t)d * 4 + h];

  const float4* fs = (const float4*)(F + (size_t)s * 768 + lane * 8);
  float u[8];
  *(float4*)&u[0] = fs[0]; *(float4*)&u[4] = fs[1];
  float* rp = rst + (size_t)d * 256 + lane * 8;
  #pragma unroll
  for (int j = 0; j < 8; ++j) atomicAdd(rp + j, alpha * u[j]);
}

// z = bf16(rst + res + bias_gat)
__global__ void k_epilogue_cvt(const float* __restrict__ rst, const float* __restrict__ F,
                               const float* __restrict__ bias_gat, __bf16* __restrict__ zb) {
  long t = (long)blockIdx.x * blockDim.x + threadIdx.x;
  if (t >= (long)N_NODES * 256) return;
  int row = (int)(t >> 8), col = (int)(t & 255);
  float z = rst[t] + F[(size_t)row * 768 + 512 + col] + bias_gat[col];
  zb[t] = (__bf16)z;
}

// ---------------- launcher ----------------

extern "C" void kernel_launch(void* const* d_in, const int* in_sizes, int n_in,
                              void* d_out, int out_size, void* d_ws, size_t ws_size,
                              hipStream_t stream) {
  const float* x        = (const float*)d_in[0];
  const int*   src      = (const int*)  d_in[1];
  const int*   dst      = (const int*)  d_in[2];
  const float* W_src    = (const float*)d_in[3];
  const float* b_src    = (const float*)d_in[4];
  const float* W_dst    = (const float*)d_in[5];
  const float* b_dst    = (const float*)d_in[6];
  const float* attn     = (const float*)d_in[7];
  const float* bias_gat = (const float*)d_in[8];
  const float* W_res    = (const float*)d_in[9];
  const float* W_dense  = (const float*)d_in[10];
  const float* b_dense  = (const float*)d_in[11];
  float* out = (float*)d_out;

  // workspace carve-up (256B aligned)
  char* base = (char*)d_ws;
  size_t off = 0;
  auto take = [&](size_t bytes) { char* p = base + off; off = (off + bytes + 255) & ~(size_t)255; return p; };
  __bf16*       xb    = (__bf16*)      take((size_t)N_NODES * IN_F * 2);   // x in bf16
  __bf16*       wt    = (__bf16*)      take((size_t)768 * IN_F * 2);       // [Wsrc|Wdst|Wres]^T bf16
  __bf16*       wdt   = (__bf16*)      take((size_t)64 * HD * 2);          // W_dense^T bf16
  float*        bcomb = (float*)       take(768 * 4);
  float*        F     = (float*)       take((size_t)N_NODES * 768 * 4);    // [fs|fd|res]
  float*        logits= (float*)       take((size_t)E_EDGES * 4 * 4);      // logits -> aexp in place
  unsigned int* mkey  = (unsigned int*)take((size_t)N_NODES * 4 * 4);
  float*        denom = (float*)       take((size_t)N_NODES * 4 * 4);
  float*        rst   = (float*)       take((size_t)N_NODES * 256 * 4);
  __bf16*       zb    = (__bf16*)      take((size_t)N_NODES * 256 * 2);
  (void)ws_size; (void)in_sizes; (void)n_in; (void)out_size;

  // 1) conversions
  {
    long n = (long)N_NODES * IN_F;
    int blocks = (int)((n + 255) / 256);
    k_cvt_bf16<<<blocks, 256, 0, stream>>>(x, xb, n);
  }
  {
    int total = IN_F * HD, blocks = (total + 255) / 256;
    k_cvt_w_t<<<blocks, 256, 0, stream>>>(W_src, wt, IN_F, HD, 0);
    k_cvt_w_t<<<blocks, 256, 0, stream>>>(W_dst, wt, IN_F, HD, 256);
    k_cvt_w_t<<<blocks, 256, 0, stream>>>(W_res, wt, IN_F, HD, 512);
  }
  {
    int total = HD * 64, blocks = (total + 255) / 256;
    k_cvt_w_t<<<blocks, 256, 0, stream>>>(W_dense, wdt, HD, 64, 0);
  }
  k_bias_combine<<<3, 256, 0, stream>>>(b_src, b_dst, bcomb);

  // 2) zero accumulators
  hipMemsetAsync(mkey,  0, (size_t)N_NODES * 4 * 4, stream);
  hipMemsetAsync(denom, 0, (size_t)N_NODES * 4 * 4, stream);
  hipMemsetAsync(rst,   0, (size_t)N_NODES * 256 * 4, stream);

  // 3) fused projection GEMM: F = x @ [Wsrc|Wdst|Wres] + [bsrc|bdst|0]
  {
    dim3 grid(768 / 64, (N_NODES + 63) / 64);
    k_gemm_bf16<<<grid, 128, 0, stream>>>(xb, wt, bcomb, F, N_NODES, 768, IN_F, 0);
  }

  // 4) edge phase
  {
    int blocks = (int)(((long)E_EDGES * 32 + 255) / 256);
    k_edge_logits<<<blocks, 256, 0, stream>>>(F, src, dst, attn, logits, mkey);
  }
  {
    int blocks = (int)(((long)E_EDGES * 4 + 255) / 256);
    k_exp_denom<<<blocks, 256, 0, stream>>>(dst, logits, mkey, denom);
  }
  {
    int blocks = (int)(((long)E_EDGES * 32 + 255) / 256);
    k_aggregate<<<blocks, 256, 0, stream>>>(F, src, dst, logits, denom, rst);
  }

  // 5) epilogue convert + final dense GEMM with ReLU
  {
    long n = (long)N_NODES * 256;
    int blocks = (int)((n + 255) / 256);
    k_epilogue_cvt<<<blocks, 256, 0, stream>>>(rst, F, bias_gat, zb);
  }
  {
    dim3 grid(64 / 64, (N_NODES + 63) / 64);
    k_gemm_bf16<<<grid, 128, 0, stream>>>(zb, wdt, b_dense, out, N_NODES, 64, HD, 1);
  }
}
